// HybridRetention_60765197304161
// MI455X (gfx1250) — matrix operand order repaired and checked by reading, not compiled, not run on hardware
//
#include <hip/hip_runtime.h>
#include <hip/hip_bf16.h>
#include <math.h>

#define DIMN  512
#define BATCH 4
#define SEQ   4096
#define ROWS  (SEQ*BATCH)   // 16384 time-major rows: row = t*BATCH + b

typedef __attribute__((ext_vector_type(16))) _Float16 v16h;
typedef __attribute__((ext_vector_type(8)))  _Float16 v8h;
typedef __attribute__((ext_vector_type(8)))  float    v8f;
typedef __attribute__((ext_vector_type(4)))  unsigned int u32x4;
typedef __attribute__((ext_vector_type(4)))  int      i32x4;
typedef __attribute__((ext_vector_type(8)))  int      i32x8;

#if __has_builtin(__builtin_amdgcn_tensor_load_to_lds)
#define HAVE_TDM 1
#else
#define HAVE_TDM 0
#endif

static __device__ __forceinline__ v16h combine16(v8h lo, v8h hi) {
  v16h r;
#pragma unroll
  for (int i = 0; i < 8; ++i) { r[i] = lo[i]; r[i + 8] = hi[i]; }
  return r;
}

static __device__ __forceinline__ v8h cvt8_f32(const float* __restrict__ p) {
  v8h r;
#pragma unroll
  for (int i = 0; i < 8; ++i) r[i] = (_Float16)p[i];
  return r;
}

#if HAVE_TDM
// Issue a TDM DMA of a contiguous BATCH x DIMN f32 tile (8 KB) from global
// memory into LDS. 2D descriptor: data_size=4B, tensor_dim0=512, tile=512x4,
// dim0_stride=512 (contiguous). Groups 2/3 zero (<=2D tensor), no multicast.
static __device__ __forceinline__ void tdm_load_rb(const float* gsrc, unsigned lds_off) {
  unsigned long long ga = (unsigned long long)(size_t)gsrc;
  u32x4 g0;
  g0[0] = 1u;                                   // count=1, user descriptor
  g0[1] = lds_off;                              // lds_addr (bytes)
  g0[2] = (unsigned)ga;                         // global_addr[31:0]
  g0[3] = (unsigned)(ga >> 32) | 0x80000000u;   // global_addr[56:32] | type=2<<30
  i32x8 g1;
  g1[0] = 0x00020000;                           // data_size=2 (4B), wg_mask=0
  g1[1] = (int)(512u << 16);                    // tensor_dim0[15:0]=512
  g1[2] = (int)(4u << 16);                      // tensor_dim0 hi=0 | tensor_dim1=4
  g1[3] = (int)(512u << 16);                    // tensor_dim1 hi=0 | tile_dim0=512
  g1[4] = 4;                                    // tile_dim1=4, tile_dim2=0
  g1[5] = 512;                                  // tensor_dim0_stride lo = 512
  g1[6] = 0;
  g1[7] = 0;
  i32x4 z4 = {0, 0, 0, 0};
#if defined(__clang_major__) && (__clang_major__ >= 23)
  i32x8 z8 = {0, 0, 0, 0, 0, 0, 0, 0};
  __builtin_amdgcn_tensor_load_to_lds(g0, g1, z4, z4, z8, 0);
#else
  __builtin_amdgcn_tensor_load_to_lds(g0, g1, z4, z4, 0);
#endif
}
#endif

// ---------------------------------------------------------------------------
// K0: convert W (512x512 f32, row-major, used as y = x @ W^T) into f16
// WMMA-B-fragment-major layout:
//   Wsw[((ntile*16 + ktile)*32 + lane)*16 + j]
//     = W[ ntile*16 + (lane&15) ][ ktile*32 + (j>>3)*16 + ((lane>>4)&1)*8 + (j&7) ]
// so each lane's 16-half B fragment is one contiguous 32-byte load.
// ---------------------------------------------------------------------------
__global__ void swizzle_wt(const float* __restrict__ W, _Float16* __restrict__ Wsw) {
  int idx = blockIdx.x * blockDim.x + threadIdx.x;   // 0 .. 512*512-1
  int j    = idx & 15;
  int lane = (idx >> 4) & 31;
  int kt   = (idx >> 9) & 15;
  int nt   = idx >> 13;
  int K = kt * 32 + (j >> 3) * 16 + ((lane >> 4) & 1) * 8 + (j & 7);
  int N = nt * 16 + (lane & 15);
  Wsw[idx] = (_Float16)W[(size_t)N * DIMN + K];
}

// ---------------------------------------------------------------------------
// K1: gate = sigmoid(q @ gate_w^T + gate_b), stored f16, time-major rows.
// One 16x16 output tile per wave; K-loop = 16 x v_wmma_f32_16x16x32_f16.
// ---------------------------------------------------------------------------
__global__ void gate_gemm(const float* __restrict__ q,
                          const _Float16* __restrict__ Gsw,
                          const float* __restrict__ gate_b,
                          _Float16* __restrict__ gatebuf) {
  const int lane = threadIdx.x & 31;
  const int wv   = threadIdx.x >> 5;
  const int mt   = blockIdx.x;
  const int nt   = blockIdx.y * 8 + wv;
  const int row  = mt * 16 + (lane & 15);
  const int t = row >> 2, b = row & 3;
  const int kb = (lane & 16) ? 8 : 0;
  const float* qrow = q + ((size_t)b * SEQ + t) * DIMN;
  v8f c = {};
#pragma unroll
  for (int kt = 0; kt < 16; ++kt) {
    const float* p = qrow + kt * 32 + kb;
    v16h a  = combine16(cvt8_f32(p), cvt8_f32(p + 16));
    v16h bm = *(const v16h*)(Gsw + (((size_t)nt * 16 + kt) * 32 + lane) * 16);
    c = __builtin_amdgcn_wmma_f32_16x16x32_f16(false, a, false, bm, (short)0, c, false, false);
  }
  const int   n     = nt * 16 + (lane & 15);
  const float bias  = gate_b[n];
  const int   mbase = (lane & 16) ? 8 : 0;
#pragma unroll
  for (int r = 0; r < 8; ++r) {
    int orow = mt * 16 + mbase + r;
    float g = 1.0f / (1.0f + __expf(-(c[r] + bias)));
    gatebuf[(size_t)orow * DIMN + n] = (_Float16)g;
  }
}

// ---------------------------------------------------------------------------
// K2: elementwise retention scan r_t = decay*r_{t-1} + k_t*v_t per channel,
// streamed out as f16 time-major rows. Coalesced over d.
// ---------------------------------------------------------------------------
__global__ void retention_scan(const float* __restrict__ k,
                               const float* __restrict__ v,
                               const float* __restrict__ decay,
                               _Float16* __restrict__ rbuf) {
  int id = blockIdx.x * blockDim.x + threadIdx.x;  // 0..2047
  int d = id & (DIMN - 1);
  int b = id >> 9;
  float dec = decay[d >> 6];
  const float* kp = k + (size_t)b * SEQ * DIMN + d;
  const float* vp = v + (size_t)b * SEQ * DIMN + d;
  float r = 0.0f;
  for (int t = 0; t < SEQ; ++t) {
    r = dec * r + kp[(size_t)t * DIMN] * vp[(size_t)t * DIMN];
    rbuf[((size_t)t * BATCH + b) * DIMN + d] = (_Float16)r;
  }
}

// ---------------------------------------------------------------------------
// K3: rB = r @ Bm^T (f16 in, f32 out), dense 16384-row WMMA GEMM.
// ---------------------------------------------------------------------------
__global__ void rb_gemm(const _Float16* __restrict__ rbuf,
                        const _Float16* __restrict__ Bsw,
                        float* __restrict__ rbbuf) {
  const int lane = threadIdx.x & 31;
  const int wv   = threadIdx.x >> 5;
  const int mt   = blockIdx.x;
  const int nt   = blockIdx.y * 8 + wv;
  const int row  = mt * 16 + (lane & 15);
  const int kb = (lane & 16) ? 8 : 0;
  const _Float16* rrow = rbuf + (size_t)row * DIMN;
  v8f c = {};
#pragma unroll
  for (int kt = 0; kt < 16; ++kt) {
    const _Float16* p = rrow + kt * 32 + kb;
    v16h a  = combine16(*(const v8h*)p, *(const v8h*)(p + 16));
    v16h bm = *(const v16h*)(Bsw + (((size_t)nt * 16 + kt) * 32 + lane) * 16);
    c = __builtin_amdgcn_wmma_f32_16x16x32_f16(false, a, false, bm, (short)0, c, false, false);
  }
  const int n     = nt * 16 + (lane & 15);
  const int mbase = (lane & 16) ? 8 : 0;
#pragma unroll
  for (int r = 0; r < 8; ++r) {
    int orow = mt * 16 + mbase + r;
    rbbuf[(size_t)orow * DIMN + n] = c[r];
  }
}

// ---------------------------------------------------------------------------
// K4: the sequential core. One workgroup, 32 waves. Wave w owns output
// columns [16w,16w+16): its 512x16 slice of A^T lives in 16 v16h registers
// (128 VGPRs) for the entire 4096-step loop. Per step:
//   blended = g*state + (1-g)*q_t  (LDS, f16, rows 0..3 of a 16-row tile)
//   c       = blended @ A^T        (16 x v_wmma per wave, B resident in regs)
//   s       = tanh(c + rB_t)       -> LDS state (f32) + global sbuf (f16)
// rB_t tiles are double-buffered in LDS via the Tensor Data Mover: wave 0
// issues tensor_load_to_lds for step t+1 during step t's WMMAs and keeps one
// DMA in flight (s_wait_tensorcnt 1; in-order TENSORcnt completion).
// ---------------------------------------------------------------------------
__global__ __launch_bounds__(1024) void seq_core(
    const float* __restrict__ q, const _Float16* __restrict__ gatebuf,
    const float* __restrict__ rbbuf, const _Float16* __restrict__ Asw,
    _Float16* __restrict__ sbuf) {
  __shared__ __align__(32) _Float16 blended[16 * DIMN];  // 16 KB
  __shared__ float state[BATCH * DIMN];                  //  8 KB
#if HAVE_TDM
  __shared__ __align__(16) float rb_lds[2 * BATCH * DIMN];  // 16 KB double buffer
#endif
  const int tid  = threadIdx.x;
  const int lane = tid & 31;
  const int wv   = tid >> 5;                // n-tile 0..31

  for (int i = tid; i < 16 * DIMN; i += 1024) blended[i] = (_Float16)0.0f;
  for (int i = tid; i < BATCH * DIMN; i += 1024) state[i] = 0.0f;

  v16h areg[16];
#pragma unroll
  for (int kt = 0; kt < 16; ++kt)
    areg[kt] = *(const v16h*)(Asw + (((size_t)wv * 16 + kt) * 32 + lane) * 16);

  const int kb    = (lane & 16) ? 8 : 0;
  const int arow  = lane & 15;
  const int n     = wv * 16 + (lane & 15);
  const int mbase = (lane & 16) ? 8 : 0;

#if HAVE_TDM
  // prologue: start DMA of rB tile for t=0 into buffer 0
  if (wv == 0)
    tdm_load_rb(rbbuf, (unsigned)(size_t)&rb_lds[0]);
#endif
  __syncthreads();

  for (int t = 0; t < SEQ; ++t) {
    // phase 1: blended = g*state + (1-g)*q_t (rows 0..3), 2 elems/thread
#pragma unroll
    for (int e = 0; e < 2; ++e) {
      int id = e * 1024 + tid;           // 0..2047
      int b = id >> 9, kk = id & (DIMN - 1);
      float g  = (float)gatebuf[((size_t)t * BATCH + b) * DIMN + kk];
      float qv = q[((size_t)b * SEQ + t) * DIMN + kk];
      float st = state[b * DIMN + kk];
      blended[b * DIMN + kk] = (_Float16)(g * st + (1.0f - g) * qv);
    }
    __syncthreads();

    // phase 2: 16x512 @ 512x16 via WMMA, weights resident in registers
    v8f c = {};
#pragma unroll
    for (int kt = 0; kt < 16; ++kt) {
      const _Float16* p = &blended[arow * DIMN + kt * 32 + kb];
      v16h a = combine16(*(const v8h*)p, *(const v8h*)(p + 16));
      c = __builtin_amdgcn_wmma_f32_16x16x32_f16(false, a, false, areg[kt], (short)0, c, false, false);
    }

#if HAVE_TDM
    // wave 0: launch DMA for step t+1, ensure step t's buffer is resident
    if (wv == 0) {
      if (t + 1 < SEQ) {
        tdm_load_rb(rbbuf + (size_t)(t + 1) * BATCH * DIMN,
                    (unsigned)(size_t)&rb_lds[((t + 1) & 1) * (BATCH * DIMN)]);
        __builtin_amdgcn_s_wait_tensorcnt(1);   // oldest (step t) complete
      } else {
        __builtin_amdgcn_s_wait_tensorcnt(0);
      }
    }
    __syncthreads();                            // publish rb_lds[t&1] to all waves
#else
    if (t + 1 < SEQ)
      __builtin_prefetch(&rbbuf[((size_t)(t + 1) * BATCH) * DIMN + n], 0, 1);
#endif

    // phase 3: s = tanh(c + rB_t); only rows m<4 are real
#pragma unroll
    for (int r = 0; r < 8; ++r) {
      int m = mbase + r;
      if (m < BATCH) {
#if HAVE_TDM
        float rbv = rb_lds[(t & 1) * (BATCH * DIMN) + m * DIMN + n];
#else
        float rbv = rbbuf[((size_t)t * BATCH + m) * DIMN + n];
#endif
        float s = tanhf(c[r] + rbv);
        state[m * DIMN + n] = s;
        sbuf[((size_t)t * BATCH + m) * DIMN + n] = (_Float16)s;
      }
    }
    __syncthreads();
  }
}

// ---------------------------------------------------------------------------
// K5: out = s @ out_w^T + out_b, written in (B,S,D) layout to d_out.
// ---------------------------------------------------------------------------
__global__ void out_gemm(const _Float16* __restrict__ sbuf,
                         const _Float16* __restrict__ Osw,
                         const float* __restrict__ out_b,
                         float* __restrict__ out) {
  const int lane = threadIdx.x & 31;
  const int wv   = threadIdx.x >> 5;
  const int mt   = blockIdx.x;
  const int nt   = blockIdx.y * 8 + wv;
  const int row  = mt * 16 + (lane & 15);
  const int kb = (lane & 16) ? 8 : 0;
  const _Float16* srow = sbuf + (size_t)row * DIMN;
  v8f c = {};
#pragma unroll
  for (int kt = 0; kt < 16; ++kt) {
    const _Float16* p = srow + kt * 32 + kb;
    v16h a  = combine16(*(const v8h*)p, *(const v8h*)(p + 16));
    v16h bm = *(const v16h*)(Osw + (((size_t)nt * 16 + kt) * 32 + lane) * 16);
    c = __builtin_amdgcn_wmma_f32_16x16x32_f16(false, a, false, bm, (short)0, c, false, false);
  }
  const int   n     = nt * 16 + (lane & 15);
  const float bias  = out_b[n];
  const int   mbase = (lane & 16) ? 8 : 0;
#pragma unroll
  for (int r = 0; r < 8; ++r) {
    int orow = mt * 16 + mbase + r;
    int t = orow >> 2, b = orow & 3;
    out[((size_t)b * SEQ + t) * DIMN + n] = c[r] + bias;
  }
}

extern "C" void kernel_launch(void* const* d_in, const int* in_sizes, int n_in,
                              void* d_out, int out_size, void* d_ws, size_t ws_size,
                              hipStream_t stream) {
  (void)in_sizes; (void)n_in; (void)out_size; (void)ws_size;
  const float* q      = (const float*)d_in[0];
  const float* k      = (const float*)d_in[1];
  const float* v      = (const float*)d_in[2];
  const float* A      = (const float*)d_in[3];
  const float* Bm     = (const float*)d_in[4];
  const float* gate_w = (const float*)d_in[5];
  const float* gate_b = (const float*)d_in[6];
  const float* out_w  = (const float*)d_in[7];
  const float* out_b  = (const float*)d_in[8];
  const float* decay  = (const float*)d_in[9];
  float* out = (float*)d_out;

  // workspace carve-up (all chunks are large power-of-two sizes -> aligned)
  char* w = (char*)d_ws;
  _Float16* Gsw = (_Float16*)w;  w += (size_t)DIMN * DIMN * sizeof(_Float16);
  _Float16* Asw = (_Float16*)w;  w += (size_t)DIMN * DIMN * sizeof(_Float16);
  _Float16* Bsw = (_Float16*)w;  w += (size_t)DIMN * DIMN * sizeof(_Float16);
  _Float16* Osw = (_Float16*)w;  w += (size_t)DIMN * DIMN * sizeof(_Float16);
  _Float16* gatebuf = (_Float16*)w;  w += (size_t)ROWS * DIMN * sizeof(_Float16);
  _Float16* rbuf    = (_Float16*)w;  w += (size_t)ROWS * DIMN * sizeof(_Float16);
  _Float16* sbuf    = (_Float16*)w;  w += (size_t)ROWS * DIMN * sizeof(_Float16);
  float*    rbbuf   = (float*)w;     w += (size_t)ROWS * DIMN * sizeof(float);

  dim3 bsw(256), gsw((DIMN * DIMN) / 256);
  swizzle_wt<<<gsw, bsw, 0, stream>>>(gate_w, Gsw);
  swizzle_wt<<<gsw, bsw, 0, stream>>>(A, Asw);
  swizzle_wt<<<gsw, bsw, 0, stream>>>(Bm, Bsw);
  swizzle_wt<<<gsw, bsw, 0, stream>>>(out_w, Osw);

  retention_scan<<<dim3(8), dim3(256), 0, stream>>>(k, v, decay, rbuf);

  dim3 ggrid(ROWS / 16, DIMN / 128), gblk(256);
  gate_gemm<<<ggrid, gblk, 0, stream>>>(q, Gsw, gate_b, gatebuf);
  rb_gemm<<<ggrid, gblk, 0, stream>>>(rbuf, Bsw, rbbuf);

  seq_core<<<dim3(1), dim3(1024), 0, stream>>>(q, gatebuf, rbbuf, Asw, sbuf);

  out_gemm<<<ggrid, gblk, 0, stream>>>(sbuf, Osw, out_b, out);
}
